// QTRAN_alt_20151986553260
// MI455X (gfx1250) — matrix operand order repaired and checked by reading, not compiled, not run on hardware
//
#include <hip/hip_runtime.h>

// QTRAN_alt counterfactual_values_hat, fused, gfx1250 WMMA (f16 in / f32 acc).
//
// Rewrites (actions are one-hot):
//   base_pre(b)  = hidden(b) @ W1_hid + sum_j W1row[j, chosen_j(b)] + b1
//   h1(b,g,a)    = relu(base_pre(b) - W1row[g, chosen_g(b)] + W1row[g, a])
// Layer-1 collapses to one K=256 GEMM + row lookups; layer-2 is the WMMA load.

typedef __attribute__((ext_vector_type(16))) _Float16 v16h;
typedef __attribute__((ext_vector_type(8)))  _Float16 h8;
typedef __attribute__((ext_vector_type(8)))  float    v8f;

#define NAG   8            // agents
#define NACT  16           // actions
#define HDIM  32           // per-agent hidden
#define HIDD  64           // MLP hidden
#define KHID  256          // NAG*HDIM: hidden-only K of the base GEMM
#define BPW   8            // batches per workgroup
#define THREADS 256        // 8 waves (wave32)
#define W1T_STRIDE 264     // padded row stride (halves) for transposed W1_hid

__launch_bounds__(THREADS)
__global__ void qtran_cf_fused(const float* __restrict__ hidden,
                               const float* __restrict__ actions,
                               const float* __restrict__ w1,
                               const float* __restrict__ b1,
                               const float* __restrict__ w2,
                               const float* __restrict__ b2,
                               const float* __restrict__ w3,
                               const float* __restrict__ b3,
                               float* __restrict__ out)
{
    __shared__ __align__(16) _Float16 s_w1t[HIDD*W1T_STRIDE]; // 33 KB: W1_hid^T [n][k]
    __shared__ __align__(16) _Float16 s_w1act[NAG*NACT*HIDD]; // 16 KB: W1 action rows
    __shared__ __align__(16) _Float16 s_w2h[HIDD*HIDD];       //  8 KB: W2 [k][n]
    __shared__ __align__(16) _Float16 s_inp[9*KHID];          // 4.5 KB: hidden rows (+1 zero row)
    __shared__ __align__(16) _Float16 s_bp[BPW*HIDD];         //  1 KB: base_pre
    __shared__ int s_ca[BPW*NAG];

    const int tid  = threadIdx.x;
    const int wave = tid >> 5;
    const int lane = tid & 31;
    const int l2   = lane & 15;
    const int half = lane >> 4;
    const int blk  = blockIdx.x;

    // ---------------- Phase 0: stage LDS (all coalesced) ----------------
    // W1 hidden columns, transposed: s_w1t[n*264 + (j*32+r)] = w1[(j*48+r)*64 + n]
    for (int i = tid; i < KHID*HIDD; i += THREADS) {
        int kh = i >> 6;                 // consecutive tid -> consecutive n
        int n  = i & 63;
        int j = kh >> 5, r = kh & 31;
        s_w1t[n*W1T_STRIDE + kh] = (_Float16)w1[(j*(HDIM+NACT) + r)*HIDD + n];
    }
    // W1 action rows (row-major in n, used by packed phase-2 math)
    for (int i = tid; i < NAG*NACT*HIDD; i += THREADS) {
        int row = i >> 6;                // g*16 + a
        int n   = i & 63;
        int g = row >> 4, a = row & 15;
        s_w1act[i] = (_Float16)w1[(g*(HDIM+NACT) + HDIM + a)*HIDD + n];
    }
    for (int i = tid; i < HIDD*HIDD; i += THREADS)
        s_w2h[i] = (_Float16)w2[i];
    // hidden rows: each batch row is 256 contiguous floats; row 8 = zeros (pad)
    for (int i = tid; i < 9*KHID; i += THREADS) {
        int b = i >> 8, kh = i & 255;
        float v = 0.f;
        if (b < BPW) v = hidden[((long)blk*BPW + b)*KHID + kh];
        s_inp[i] = (_Float16)v;
    }
    if (tid < BPW*NAG) {
        int b = tid >> 3, g = tid & 7;
        const float* ap = actions + (((long)blk*BPW + b)*NAG + g)*NACT;
        int c = 0;
        #pragma unroll
        for (int a = 0; a < NACT; ++a) if (ap[a] > 0.5f) c = a;
        s_ca[tid] = c;
    }
    __syncthreads();

    // -------- Phase 1: hidden @ W1_hid + b1 (waves 0..3, one N-tile each) --------
    if (wave < 4) {
        const int nt = wave;
        const int n  = nt*16 + l2;
        const float bias = b1[n];
        v8f c;
        #pragma unroll
        for (int r = 0; r < 8; ++r) c[r] = bias;
        const int rr = (l2 < BPW) ? l2 : BPW;          // pad lanes read the zero row
        const _Float16* inr = s_inp + rr*KHID;
        const _Float16* wtr = s_w1t + n*W1T_STRIDE;
        #pragma unroll
        for (int ks = 0; ks < KHID/32; ++ks) {
            // A-frag: two contiguous 8-half runs (ISA 16-bit A 16x32 layout)
            int o0 = ks*32 + half*8;
            h8 x0 = *(const h8*)(inr + o0);
            h8 x1 = *(const h8*)(inr + o0 + 16);
            v16h aF = __builtin_shufflevector(x0, x1,
                         0,1,2,3,4,5,6,7,8,9,10,11,12,13,14,15);
            // B-frag: two 8-half runs at K = ks*32 + 8*half and +16 (matches A) in transposed W1
            int ob = ks*32 + half*8;
            h8 y0 = *(const h8*)(wtr + ob);
            h8 y1 = *(const h8*)(wtr + ob + 16);
            v16h bF = __builtin_shufflevector(y0, y1,
                         0,1,2,3,4,5,6,7,8,9,10,11,12,13,14,15);
            c = __builtin_amdgcn_wmma_f32_16x16x32_f16(false, aF, false, bF,
                                                       (short)0, c, false, false);
        }
        if (half == 0) {    // rows M=0..7 are the valid batches
            #pragma unroll
            for (int r = 0; r < 8; ++r) s_bp[r*HIDD + n] = (_Float16)c[r];
        }
    }
    __syncthreads();

    // Add the one-hot action contributions: bp += sum_j W1row[j, chosen_j]
    for (int i = tid; i < BPW*HIDD; i += THREADS) {
        int b = i >> 6, n = i & 63;
        float acc = (float)s_bp[i];
        #pragma unroll
        for (int j = 0; j < NAG; ++j)
            acc += (float)s_w1act[(j*NACT + s_ca[b*NAG + j])*HIDD + n];
        s_bp[i] = (_Float16)acc;
    }
    __syncthreads();

    // -------- Phase 2: one (b,g) tile = 16 action rows = one WMMA M-tile --------
    // Hoist all W2 B-fragments (K=64 -> 2 chunks, N=64 -> 4 tiles): full reuse.
    v16h bw2[4][2];
    #pragma unroll
    for (int nt = 0; nt < 4; ++nt)
        #pragma unroll
        for (int cc = 0; cc < 2; ++cc)
            #pragma unroll
            for (int e = 0; e < 16; ++e)
                bw2[nt][cc][e] = s_w2h[(cc*32 + half*8 + e + (e>>3)*8)*HIDD + nt*16 + l2];

    float bias2[4], w3v[4];
    #pragma unroll
    for (int nt = 0; nt < 4; ++nt) {
        bias2[nt] = b2[nt*16 + l2];
        w3v[nt]   = w3[nt*16 + l2];
    }
    const float b3s = b3[0];
    const h8 zero = {};

    const int TILES = BPW*NAG;   // 64 tiles, 8 per wave
    for (int t = wave; t < TILES; t += THREADS/32) {
        const int b  = t >> 3, g = t & 7;
        const int ca = s_ca[t];
        const _Float16* bp  = s_bp + b*HIDD;
        const _Float16* wg  = s_w1act + (g*NACT + l2)*HIDD;  // this lane's action row
        const _Float16* wca = s_w1act + (g*NACT + ca)*HIDD;  // chosen-action row

        // Layer-1 output built straight into A-fragment layout with packed f16 math
        v16h afrag[2];
        #pragma unroll
        for (int cc = 0; cc < 2; ++cc) {
            int o0 = cc*32 + half*8;
            int o1 = o0 + 16;
            h8 x0 = *(const h8*)(bp + o0) - *(const h8*)(wca + o0) + *(const h8*)(wg + o0);
            h8 x1 = *(const h8*)(bp + o1) - *(const h8*)(wca + o1) + *(const h8*)(wg + o1);
            x0 = __builtin_elementwise_max(x0, zero);   // v_pk_max_f16 (ReLU)
            x1 = __builtin_elementwise_max(x1, zero);
            afrag[cc] = __builtin_shufflevector(x0, x1,
                           0,1,2,3,4,5,6,7,8,9,10,11,12,13,14,15);
        }

        v8f q;
        #pragma unroll
        for (int r = 0; r < 8; ++r) q[r] = 0.f;

        #pragma unroll
        for (int nt = 0; nt < 4; ++nt) {
            v8f c;
            #pragma unroll
            for (int r = 0; r < 8; ++r) c[r] = bias2[nt];
            c = __builtin_amdgcn_wmma_f32_16x16x32_f16(false, afrag[0], false, bw2[nt][0],
                                                       (short)0, c, false, false);
            c = __builtin_amdgcn_wmma_f32_16x16x32_f16(false, afrag[1], false, bw2[nt][1],
                                                       (short)0, c, false, false);
            // layer 3: relu(h2) . w3, each lane owns column nt*16+l2
            #pragma unroll
            for (int r = 0; r < 8; ++r) q[r] += fmaxf(c[r], 0.f) * w3v[nt];
        }

        // reduce over the 16-lane N groups (xor stays within each half)
        #pragma unroll
        for (int m = 1; m <= 8; m <<= 1)
            #pragma unroll
            for (int r = 0; r < 8; ++r) q[r] += __shfl_xor(q[r], m, 32);

        if (l2 == 0) {  // lane0 -> actions 0..7, lane16 -> actions 8..15
            long bb = (long)blk*BPW + b;
            float* op = out + (bb*NAG + g)*NACT + half*8;
            float4 o0 = make_float4(q[0]+b3s, q[1]+b3s, q[2]+b3s, q[3]+b3s);
            float4 o1 = make_float4(q[4]+b3s, q[5]+b3s, q[6]+b3s, q[7]+b3s);
            *(float4*)op       = o0;
            *((float4*)op + 1) = o1;
        }
    }
}

extern "C" void kernel_launch(void* const* d_in, const int* in_sizes, int n_in,
                              void* d_out, int out_size, void* d_ws, size_t ws_size,
                              hipStream_t stream) {
    const float* hidden  = (const float*)d_in[0];
    const float* actions = (const float*)d_in[1];
    const float* w1 = (const float*)d_in[2];
    const float* b1 = (const float*)d_in[3];
    const float* w2 = (const float*)d_in[4];
    const float* b2 = (const float*)d_in[5];
    const float* w3 = (const float*)d_in[6];
    const float* b3 = (const float*)d_in[7];
    float* out = (float*)d_out;

    const int B = in_sizes[0] / (NAG*HDIM);     // 2048
    const int grid = (B + BPW - 1) / BPW;       // 256 workgroups
    qtran_cf_fused<<<grid, THREADS, 0, stream>>>(hidden, actions, w1, b1,
                                                 w2, b2, w3, b3, out);
}
